// parallel_log_minGRU_58789512347903
// MI455X (gfx1250) — compile-verified, matrix operand order checked by reading, not run on hardware
//
#include <hip/hip_runtime.h>
#include <hip/hip_bf16.h>
#include <math.h>

#define Bsz    8
#define Tlen   4096
#define DIN    1024
#define DH     1024
#define MTOT   (Bsz * Tlen)      // 32768
#define CHUNKS 16
#define CLEN   (Tlen / CHUNKS)   // 256

typedef __attribute__((ext_vector_type(16))) __bf16         v16bf;
typedef __attribute__((ext_vector_type(16))) unsigned short v16us;
typedef __attribute__((ext_vector_type(8)))  unsigned short v8us;
typedef __attribute__((ext_vector_type(8)))  float          v8f;
typedef __attribute__((ext_vector_type(4)))  float          f32x4;

// ---------- helpers ----------
__device__ __forceinline__ unsigned short f2bf(float f) {
  unsigned int u = __builtin_bit_cast(unsigned int, f);
  u += 0x7FFFu + ((u >> 16) & 1u);          // round-to-nearest-even
  return (unsigned short)(u >> 16);
}

// Load one 16-bit A/B WMMA fragment (ISA 7.12.2 layout) from a pre-converted
// bf16 (ushort) row-major matrix. p points at row_base + k0 + laneKOff where
// laneKOff = (lane>=16 ? 8 : 0). Fragment element i -> k = (i<8 ? i : i+8)+off,
// i.e. bf16 elements [0..7] and [16..23] from p: two 16-byte loads.
__device__ __forceinline__ v16bf load_frag16(const unsigned short* __restrict__ p) {
  const v8us* q = (const v8us*)p;
  v8us lo = q[0], hi = q[2];
  return __builtin_bit_cast(
      v16bf, __builtin_shufflevector(lo, hi, 0, 1, 2, 3, 4, 5, 6, 7,
                                     8, 9, 10, 11, 12, 13, 14, 15));
}

__device__ __forceinline__ float softplusf(float x) {
  return fmaxf(x, 0.0f) + log1pf(__expf(-fabsf(x)));
}
__device__ __forceinline__ float log_g(float x) {
  return (x >= 0.0f) ? __logf(x + 0.5f) : -softplusf(-x);
}
__device__ __forceinline__ float logaddexpf_(float x, float y) {
  float mx = fmaxf(x, y), mn = fminf(x, y);
  if (mx <= -3.0e38f) return mx;            // both ~ -inf
  return mx + log1pf(__expf(mn - mx));
}

#define WMMA_BF16(A, Bf, C) \
  __builtin_amdgcn_wmma_f32_16x16x32_bf16(false, (A), false, (Bf), (short)0, (C), false, false)

// ---------- kernel 0: one-shot f32 -> bf16 conversion (memory-bound) ----------
__global__ __launch_bounds__(256) void cvt_f32_to_bf16(
    const float* __restrict__ src, unsigned short* __restrict__ dst, int n8) {
  const int i = blockIdx.x * blockDim.x + threadIdx.x;
  if (i >= n8) return;
  const f32x4* s4 = (const f32x4*)src;
  f32x4 a = s4[2 * i], b = s4[2 * i + 1];
  v8us u;
  u[0] = f2bf(a.x); u[1] = f2bf(a.y); u[2] = f2bf(a.z); u[3] = f2bf(a.w);
  u[4] = f2bf(b.x); u[5] = f2bf(b.y); u[6] = f2bf(b.z); u[7] = f2bf(b.w);
  ((v8us*)dst)[i] = u;
}

// ---------- kernel 1: dual GEMM (bf16 WMMA) + fused log-space epilogue ----------
// Block = 256 threads (8 waves, arranged 4M x 2N). Block tile: 128(M) x 64(N).
// Wave tile: 32(M) x 32(N) for BOTH Wz and Wh => 8 accumulators, and per K-step
// only 2 A-fragments + 4 B-fragments feed 8 WMMAs (6 frag loads / 8 WMMA).
__global__ __launch_bounds__(256) void mingru_proj_wmma(
    const unsigned short* __restrict__ xb,  const unsigned short* __restrict__ wzb,
    const unsigned short* __restrict__ whb,
    const float* __restrict__ bz, const float* __restrict__ bh,
    float* __restrict__ logc,     // -> log(1 - sigmoid(z_pre))     [MTOT, DH]
    float* __restrict__ logv) {   // -> log(sigmoid)+log_g(h_pre)   [MTOT, DH]
  const int lane  = threadIdx.x & 31;
  const int wave  = threadIdx.x >> 5;
  const int r     = lane & 15;          // fragment row/col within 16
  const int hi8   = (lane >> 4) << 3;   // 0 or 8: K-half for frags, M-half for C/D
  const int waveM = wave & 3, waveN = wave >> 2;
  const int mBase = blockIdx.x * 128 + waveM * 32;
  const int nBase = blockIdx.y * 64 + waveN * 32;

  v8f accZ[2][2], accH[2][2];           // [m-tile][n-tile]
#pragma unroll
  for (int mt = 0; mt < 2; ++mt)
#pragma unroll
    for (int nt = 0; nt < 2; ++nt)
#pragma unroll
      for (int i = 0; i < 8; ++i) { accZ[mt][nt][i] = 0.0f; accH[mt][nt][i] = 0.0f; }

  const unsigned short* aRow0  = xb  + (size_t)(mBase + r) * DIN;
  const unsigned short* aRow1  = aRow0 + (size_t)16 * DIN;
  const unsigned short* wzRow0 = wzb + (size_t)(nBase + r) * DIN;
  const unsigned short* wzRow1 = wzRow0 + (size_t)16 * DIN;
  const unsigned short* whRow0 = whb + (size_t)(nBase + r) * DIN;
  const unsigned short* whRow1 = whRow0 + (size_t)16 * DIN;

  for (int k0 = 0; k0 < DIN; k0 += 32) {
    const int o = k0 + hi8;
    __builtin_prefetch(aRow0 + k0 + 128, 0, 0);        // global_prefetch_b8

    v16bf a0 = load_frag16(aRow0 + o);
    v16bf a1 = load_frag16(aRow1 + o);

    v16bf b0 = load_frag16(wzRow0 + o);
    accZ[0][0] = WMMA_BF16(a0, b0, accZ[0][0]);
    accZ[1][0] = WMMA_BF16(a1, b0, accZ[1][0]);
    v16bf b1 = load_frag16(wzRow1 + o);
    accZ[0][1] = WMMA_BF16(a0, b1, accZ[0][1]);
    accZ[1][1] = WMMA_BF16(a1, b1, accZ[1][1]);

    v16bf c0 = load_frag16(whRow0 + o);
    accH[0][0] = WMMA_BF16(a0, c0, accH[0][0]);
    accH[1][0] = WMMA_BF16(a1, c0, accH[1][0]);
    v16bf c1 = load_frag16(whRow1 + o);
    accH[0][1] = WMMA_BF16(a0, c1, accH[0][1]);
    accH[1][1] = WMMA_BF16(a1, c1, accH[1][1]);
  }

  // C/D layout: VGPR v holds M = v (lanes 0-15) or M = v+8 (lanes 16-31), N = lane%16.
#pragma unroll
  for (int mt = 0; mt < 2; ++mt) {
    const int rowBase = mBase + mt * 16 + hi8;
#pragma unroll
    for (int nt = 0; nt < 2; ++nt) {
      const int col = nBase + nt * 16 + r;
      const float bzv = bz[col], bhv = bh[col];
#pragma unroll
      for (int v = 0; v < 8; ++v) {
        const size_t oo = (size_t)(rowBase + v) * DH + col;
        float zp = accZ[mt][nt][v] + bzv;
        float hp = accH[mt][nt][v] + bhv;
        logc[oo] = -softplusf(zp);                     // log(1 - sigmoid(zp))
        logv[oo] = -softplusf(-zp) + log_g(hp);        // log(sigmoid(zp)) + log g(hp)
      }
    }
  }
}

// ---------- kernel 2: per-chunk local scan ----------
__global__ __launch_bounds__(256) void scan_pass1(
    const float* __restrict__ logc, const float* __restrict__ logv,
    float* __restrict__ atot, float* __restrict__ mv) {
  const int idx = blockIdx.x * blockDim.x + threadIdx.x;  // (b*CHUNKS+c)*DH + d
  const int d = idx % DH;
  const int bc = idx / DH;
  const int c = bc % CHUNKS;
  const int b = bc / CHUNKS;
  size_t base = ((size_t)b * Tlen + (size_t)c * CLEN) * DH + d;
  float A = 0.0f, M = -INFINITY;
  for (int t = 0; t < CLEN; ++t) {
    float a = logc[base], v = logv[base];
    A += a;
    M = logaddexpf_(a + M, v);
    base += DH;
  }
  atot[idx] = A;
  mv[idx]   = M;
}

// ---------- kernel 3: carry propagation across chunks ----------
__global__ __launch_bounds__(256) void scan_pass2(
    const float* __restrict__ h_prev, const float* __restrict__ atot,
    const float* __restrict__ mv, float* __restrict__ carry) {
  const int idx = blockIdx.x * blockDim.x + threadIdx.x;  // b*DH + d
  const int d = idx % DH;
  const int b = idx / DH;
  float c = log_g(h_prev[idx]);
  for (int cc = 0; cc < CHUNKS; ++cc) {
    const size_t o = ((size_t)b * CHUNKS + cc) * DH + d;
    carry[o] = c;
    c = logaddexpf_(atot[o] + c, mv[o]);
  }
}

// ---------- kernel 4: replay with carry, write exp(log_h) in place ----------
__global__ __launch_bounds__(256) void scan_pass3(
    const float* __restrict__ logc, const float* __restrict__ carry,
    float* __restrict__ out) {  // out holds logv on entry, h on exit
  const int idx = blockIdx.x * blockDim.x + threadIdx.x;
  const int d = idx % DH;
  const int bc = idx / DH;
  const int c = bc % CHUNKS;
  const int b = bc / CHUNKS;
  size_t base = ((size_t)b * Tlen + (size_t)c * CLEN) * DH + d;
  float L = carry[idx];
  for (int t = 0; t < CLEN; ++t) {
    float a = logc[base], v = out[base];   // read before in-place write
    L = logaddexpf_(a + L, v);
    out[base] = __expf(L);
    base += DH;
  }
}

extern "C" void kernel_launch(void* const* d_in, const int* in_sizes, int n_in,
                              void* d_out, int out_size, void* d_ws, size_t ws_size,
                              hipStream_t stream) {
  const float* x      = (const float*)d_in[0];
  const float* h_prev = (const float*)d_in[1];
  const float* Wz     = (const float*)d_in[2];
  const float* bz     = (const float*)d_in[3];
  const float* Wh     = (const float*)d_in[4];
  const float* bh     = (const float*)d_in[5];
  float* out = (float*)d_out;

  // workspace layout (all offsets 16B-aligned):
  unsigned short* xb  = (unsigned short*)d_ws;            // MTOT*DIN bf16 (64 MB)
  unsigned short* wzb = xb + (size_t)MTOT * DIN;          // DH*DIN bf16  (2 MB)
  unsigned short* whb = wzb + (size_t)DH * DIN;           // DH*DIN bf16  (2 MB)
  float* logc  = (float*)(whb + (size_t)DH * DIN);        // MTOT*DH f32  (128 MB)
  float* atot  = logc + (size_t)MTOT * DH;                // Bsz*CHUNKS*DH
  float* mv    = atot + (size_t)Bsz * CHUNKS * DH;
  float* carry = mv   + (size_t)Bsz * CHUNKS * DH;

  // 0) one-shot bf16 conversion of x, Wz, Wh
  {
    const int n8x = MTOT * DIN / 8;   // 4194304
    cvt_f32_to_bf16<<<n8x / 256, 256, 0, stream>>>(x, xb, n8x);
    const int n8w = DH * DIN / 8;     // 131072
    cvt_f32_to_bf16<<<n8w / 256, 256, 0, stream>>>(Wz, wzb, n8w);
    cvt_f32_to_bf16<<<n8w / 256, 256, 0, stream>>>(Wh, whb, n8w);
  }

  // 1) dual WMMA GEMM + fused log-space epilogue
  dim3 gGemm(MTOT / 128, DH / 64);
  mingru_proj_wmma<<<gGemm, 256, 0, stream>>>(xb, wzb, whb, bz, bh, logc, out);

  // 2-4) chunked log-space scan over T
  const int n1 = Bsz * CHUNKS * DH;  // 131072
  scan_pass1<<<n1 / 256, 256, 0, stream>>>(logc, out, atot, mv);
  const int n2 = Bsz * DH;           // 8192
  scan_pass2<<<n2 / 256, 256, 0, stream>>>(h_prev, atot, mv, carry);
  scan_pass3<<<n1 / 256, 256, 0, stream>>>(logc, carry, out);
}
